// STBlock_24515673325955
// MI455X (gfx1250) — matrix operand order repaired and checked by reading, not compiled, not run on hardware
//
#include <hip/hip_runtime.h>

// ---------------------------------------------------------------------------
// ST-GCN block for MI455X (gfx1250, wave32, WMMA).
// Dominant cost: 3x dense GEMM  S[8192x768] = A[8192x8192] @ X[8192x768].
// bf16 hi/lo split (3-term) with fp32 accumulate on v_wmma_f32_16x16x32_bf16.
// GEMM: 128x128x64 tiles, register double-buffering of the next tile so global
// latency overlaps WMMA; column-block-fastest grid so A row-blocks are served
// from the 192MB L2. If workspace allows (~352MB), A is pre-converted to bf16
// hi/lo once, removing the 18x redundant fp32->bf16 split VALU work.
// ---------------------------------------------------------------------------

typedef __bf16 bf16_t;
typedef __attribute__((ext_vector_type(4)))  float  v4f;
typedef __attribute__((ext_vector_type(8)))  float  v8f;
typedef __attribute__((ext_vector_type(4)))  bf16_t v4bf;
typedef __attribute__((ext_vector_type(8)))  bf16_t v8bf;
typedef __attribute__((ext_vector_type(16))) bf16_t v16bf;

#define NN   8192   // nodes
#define TT   12     // timesteps
#define FF   64     // feature width (F_IN == H1 == H2 == F_OUT == 64)
#define CC   768    // TT*FF columns of the aggregation GEMM
#define RR   98304  // NN*TT rows of the per-node transforms

__device__ __forceinline__ void split_f32(float x, bf16_t& h, bf16_t& l) {
    h = (bf16_t)x;
    l = (bf16_t)(x - (float)h);
}

__device__ __forceinline__ v16bf pack16(v8bf a, v8bf b) {
    v16bf r;
#pragma unroll
    for (int i = 0; i < 8; ++i) { r[i] = a[i]; r[i + 8] = b[i]; }
    return r;
}

// A fragment (16x32 bf16, row-major tile with row stride ldk elements).
// ISA layout: lanes 0-15 hold M=row, K={0..7,16..23}; lanes 16-31 K={8..15,24..31}.
__device__ __forceinline__ v16bf load_a_frag(const bf16_t* tile, int ldk) {
    const int lane = threadIdx.x & 31;
    const int row  = lane & 15;
    const int k0   = (lane < 16) ? 0 : 8;
    const bf16_t* p = tile + row * ldk + k0;
    return pack16(*(const v8bf*)p, *(const v8bf*)(p + 16));
}

// B fragment (32x16 bf16) from a [col][k] tile with col stride ldk elements.
// Lanes 0-15: col=lane, K={0..7,16..23} ; lanes 16-31: col=lane-16, K={8..15,24..31}.
__device__ __forceinline__ v16bf load_b_frag(const bf16_t* tile, int ldk) {
    const int lane = threadIdx.x & 31;
    const int col  = lane & 15;
    const int k0   = (lane < 16) ? 0 : 8;
    const bf16_t* p = tile + col * ldk + k0;
    return pack16(*(const v8bf*)p, *(const v8bf*)(p + 16));
}

// 3-term split-precision MAC: C += Ah*Bh + Ah*Bl + Al*Bh (fp32 accumulate)
__device__ __forceinline__ v8f wmma_split(v16bf ah, v16bf al, v16bf bh, v16bf bl, v8f c) {
    c = __builtin_amdgcn_wmma_f32_16x16x32_bf16(false, ah, false, bh, (short)0, c, false, false);
    c = __builtin_amdgcn_wmma_f32_16x16x32_bf16(false, ah, false, bl, (short)0, c, false, false);
    c = __builtin_amdgcn_wmma_f32_16x16x32_bf16(false, al, false, bh, (short)0, c, false, false);
    return c;
}

// ---------------------------------------------------------------------------
// Kernel 0: one-time A fp32 -> bf16 hi/lo (big-workspace path)
// ---------------------------------------------------------------------------
__global__ __launch_bounds__(256) void convert_a_kernel(const float* __restrict__ A,
                                                        bf16_t* __restrict__ Ah,
                                                        bf16_t* __restrict__ Al) {
    const size_t gid = ((size_t)blockIdx.x * 256 + threadIdx.x) * 4;
    v4f v = *(const v4f*)(A + gid);
    v4bf h4, l4;
#pragma unroll
    for (int j = 0; j < 4; ++j) {
        bf16_t h, l;
        split_f32(v[j], h, l);
        h4[j] = h; l4[j] = l;
    }
    *(v4bf*)(Ah + gid) = h4;
    *(v4bf*)(Al + gid) = l4;
}

// ---------------------------------------------------------------------------
// Kernel 1: pack temporal_features [N][F][T] fp32 -> X^T hi/lo [768][8192] bf16
// column index c = t*64 + f ; element X^T[c][n] = tf[n][f][t]
// ---------------------------------------------------------------------------
__global__ __launch_bounds__(256) void pack_kernel(const float* __restrict__ tf,
                                                   bf16_t* __restrict__ Xth,
                                                   bf16_t* __restrict__ Xtl) {
    const long gid = (long)blockIdx.x * 256 + threadIdx.x;
    if (gid >= (long)CC * NN) return;
    const int n = (int)(gid & (NN - 1));
    const int c = (int)(gid >> 13);
    const int f = c & 63;
    const int t = c >> 6;
    const float x = tf[(size_t)n * (FF * TT) + f * TT + t];
    bf16_t h, l;
    split_f32(x, h, l);
    Xth[gid] = h;
    Xtl[gid] = l;
}

// ---------------------------------------------------------------------------
// Kernel 2: S[8192x768] = A[8192x8192] @ X (X as bf16 hi/lo X^T [768][8192])
// BM=128, BN=128, BK=64; 256 threads = 8 waves; wave grid 4(M) x 2(N);
// each wave owns 2x4 accumulator tiles; next tile prefetched into VGPRs while
// the current LDS tile feeds 48 WMMAs per wave. ABF16 selects pre-converted A.
// ---------------------------------------------------------------------------
#define BM 128
#define BN 128
#define BK 64

template <bool ABF16>
__global__ __launch_bounds__(256) void gemm_adj_kernel(const float*  __restrict__ A,
                                                       const bf16_t* __restrict__ Agh,
                                                       const bf16_t* __restrict__ Agl,
                                                       const bf16_t* __restrict__ Bth,
                                                       const bf16_t* __restrict__ Btl,
                                                       float* __restrict__ S) {
    __shared__ bf16_t Ash[BM * BK];   // 16KB each; 64KB total (<< 320KB/WGP)
    __shared__ bf16_t Asl[BM * BK];
    __shared__ bf16_t Bsh[BN * BK];
    __shared__ bf16_t Bsl[BN * BK];

    const int tid  = threadIdx.x;
    const int wave = tid >> 5;
    const int nBase = blockIdx.x * BN;   // column block fastest -> L2 reuse of A rows
    const int mBase = blockIdx.y * BM;
    const int wm = wave & 3;             // 4 waves along M (32 rows each)
    const int wn = wave >> 2;            // 2 waves along N (64 cols each)

    v8f acc[2][4] = {};

    const int srow = tid >> 1;           // 0..127 staged row
    const int skk  = (tid & 1) * 32;     // 0 / 32 staged k-half

    const float*  apF = A   + (size_t)(mBase + srow) * NN + skk;   // fp32 mode
    const bf16_t* apH = Agh + (size_t)(mBase + srow) * NN + skk;   // bf16 mode
    const bf16_t* apL = Agl + (size_t)(mBase + srow) * NN + skk;
    const bf16_t* bpH = Bth + (size_t)(nBase + srow) * NN + skk;
    const bf16_t* bpL = Btl + (size_t)(nBase + srow) * NN + skk;

    // staging registers (double buffer vs LDS)
    v4f  aF[8];
    v8bf aH[4], aL[4];
    v8bf bH[4], bL[4];

    auto load_tile = [&](int k0) {
        if (ABF16) {
#pragma unroll
            for (int i = 0; i < 4; ++i) {
                aH[i] = *(const v8bf*)(apH + k0 + 8 * i);
                aL[i] = *(const v8bf*)(apL + k0 + 8 * i);
            }
        } else {
#pragma unroll
            for (int i = 0; i < 8; ++i) aF[i] = *(const v4f*)(apF + k0 + 4 * i);
        }
#pragma unroll
        for (int i = 0; i < 4; ++i) {
            bH[i] = *(const v8bf*)(bpH + k0 + 8 * i);
            bL[i] = *(const v8bf*)(bpL + k0 + 8 * i);
        }
    };

    auto store_tile = [&]() {
        bf16_t* dAh = &Ash[srow * BK + skk];
        bf16_t* dAl = &Asl[srow * BK + skk];
        if (ABF16) {
#pragma unroll
            for (int i = 0; i < 4; ++i) {
                *(v8bf*)(dAh + 8 * i) = aH[i];
                *(v8bf*)(dAl + 8 * i) = aL[i];
            }
        } else {
#pragma unroll
            for (int g = 0; g < 4; ++g) {
                v8bf h8, l8;
#pragma unroll
                for (int j = 0; j < 8; ++j) {
                    bf16_t h, l;
                    split_f32(aF[g * 2 + (j >> 2)][j & 3], h, l);
                    h8[j] = h; l8[j] = l;
                }
                *(v8bf*)(dAh + 8 * g) = h8;
                *(v8bf*)(dAl + 8 * g) = l8;
            }
        }
        bf16_t* dBh = &Bsh[srow * BK + skk];
        bf16_t* dBl = &Bsl[srow * BK + skk];
#pragma unroll
        for (int i = 0; i < 4; ++i) {
            *(v8bf*)(dBh + 8 * i) = bH[i];
            *(v8bf*)(dBl + 8 * i) = bL[i];
        }
    };

    load_tile(0);
    for (int k0 = 0; k0 < NN; k0 += BK) {
        store_tile();
        __syncthreads();
        if (k0 + BK < NN) load_tile(k0 + BK);   // overlaps with compute below

#pragma unroll
        for (int ks = 0; ks < 2; ++ks) {
            const int kk = ks * 32;
            v16bf fah[2], fal[2];
#pragma unroll
            for (int mi = 0; mi < 2; ++mi) {
                const int rb = (wm * 2 + mi) * 16;
                fah[mi] = load_a_frag(&Ash[rb * BK] + kk, BK);
                fal[mi] = load_a_frag(&Asl[rb * BK] + kk, BK);
            }
#pragma unroll
            for (int ni = 0; ni < 4; ++ni) {
                const int cb = (wn * 4 + ni) * 16;
                v16bf fbh = load_b_frag(&Bsh[cb * BK] + kk, BK);
                v16bf fbl = load_b_frag(&Bsl[cb * BK] + kk, BK);
#pragma unroll
                for (int mi = 0; mi < 2; ++mi)
                    acc[mi][ni] = wmma_split(fah[mi], fal[mi], fbh, fbl, acc[mi][ni]);
            }
        }
        __syncthreads();
    }

    // epilogue: fp32 store (C/D layout: vgpr v, lane l -> M=v+8*(l/16), N=l%16)
    const int lane = tid & 31;
    const int ln = lane & 15;
    const int lh = lane >> 4;
#pragma unroll
    for (int mi = 0; mi < 2; ++mi) {
#pragma unroll
        for (int ni = 0; ni < 4; ++ni) {
            const int col  = nBase + (wn * 4 + ni) * 16 + ln;
            const int rowb = mBase + (wm * 2 + mi) * 16 + lh * 8;
#pragma unroll
            for (int v = 0; v < 8; ++v)
                S[(size_t)(rowb + v) * CC + col] = acc[mi][ni][v];
        }
    }
}

// ---------------------------------------------------------------------------
// Kernel 3: per-node feature transform  Y = act(S @ W + b)
// Grid: (t = 0..11, nodeBlock = 0..63); block covers 128 nodes x 64 outputs.
// out_transposed=1: write bf16 hi/lo X^T [768][8192] (input for next GEMM)
// out_transposed=0: write bf16 hi/lo row-major [98304][64] (input for conv)
// ---------------------------------------------------------------------------
__global__ __launch_bounds__(256) void transform_kernel(const float* __restrict__ S,
                                                        const float* __restrict__ W,
                                                        const float* __restrict__ bias,
                                                        int do_relu, int out_transposed,
                                                        bf16_t* __restrict__ outTh,
                                                        bf16_t* __restrict__ outTl,
                                                        bf16_t* __restrict__ outRh,
                                                        bf16_t* __restrict__ outRl) {
    __shared__ __align__(16) char smem[49152];
    bf16_t* Ssh = (bf16_t*)smem;              // [128][64]  16KB
    bf16_t* Ssl = (bf16_t*)(smem + 16384);    // [128][64]  16KB
    bf16_t* Wsh = (bf16_t*)(smem + 32768);    // [o][f]      8KB
    bf16_t* Wsl = (bf16_t*)(smem + 40960);    // [o][f]      8KB
    float*  accS = (float*)smem;              // [128][65] aliased after compute

    const int t     = blockIdx.x;
    const int nBase = blockIdx.y * 128;
    const int tid   = threadIdx.x;
    const int wave  = tid >> 5;
    const int lane  = tid & 31;

    // stage S rows (fp32 -> bf16 hi/lo): row = node-local, 64 contiguous f
    {
        const int row = tid >> 1;
        const int fh  = (tid & 1) * 32;
        const float* sp = S + ((size_t)(nBase + row) * TT + t) * FF + fh;
#pragma unroll
        for (int i = 0; i < 32; i += 4) {
            v4f v = *(const v4f*)(sp + i);
#pragma unroll
            for (int j = 0; j < 4; ++j) {
                bf16_t h, l;
                split_f32(v[j], h, l);
                Ssh[row * FF + fh + i + j] = h;
                Ssl[row * FF + fh + i + j] = l;
            }
        }
    }
    // stage W transposed into [o][f]
    {
        const int o  = tid >> 2;
        const int f0 = (tid & 3) * 16;
#pragma unroll
        for (int j = 0; j < 16; ++j) {
            bf16_t h, l;
            split_f32(W[(size_t)(f0 + j) * FF + o], h, l);
            Wsh[o * FF + f0 + j] = h;
            Wsl[o * FF + f0 + j] = l;
        }
    }
    __syncthreads();

    // compute: wave handles rows wave*16..+15, all 64 output cols
    v8f acc[4] = {};
#pragma unroll
    for (int kk = 0; kk < FF; kk += 32) {
        v16bf ah = load_a_frag(&Ssh[(wave * 16) * FF] + kk, FF);
        v16bf al = load_a_frag(&Ssl[(wave * 16) * FF] + kk, FF);
#pragma unroll
        for (int ni = 0; ni < 4; ++ni) {
            v16bf bh = load_b_frag(&Wsh[(ni * 16) * FF] + kk, FF);
            v16bf bl = load_b_frag(&Wsl[(ni * 16) * FF] + kk, FF);
            acc[ni] = wmma_split(ah, al, bh, bl, acc[ni]);
        }
    }
    __syncthreads();   // all LDS reads done before aliasing accS over the tiles

    // bias + activation -> accS [128][65]
    {
        const int ln = lane & 15;
        const int lh = lane >> 4;
#pragma unroll
        for (int ni = 0; ni < 4; ++ni) {
            const int col = ni * 16 + ln;
            const float b = bias[col];
#pragma unroll
            for (int v = 0; v < 8; ++v) {
                float x = acc[ni][v] + b;
                if (do_relu) x = x > 0.0f ? x : 0.0f;
                accS[(wave * 16 + lh * 8 + v) * 65 + col] = x;
            }
        }
    }
    __syncthreads();

    if (out_transposed) {
        // column o of X^T: 32 consecutive nodes per thread -> 64B bf16 stores
        const int o  = tid >> 2;
        const int s0 = (tid & 3) * 32;
        const size_t base = (size_t)(t * FF + o) * NN + nBase + s0;
#pragma unroll
        for (int j0 = 0; j0 < 32; j0 += 8) {
            v8bf vh, vl;
#pragma unroll
            for (int j = 0; j < 8; ++j) {
                bf16_t h, l;
                split_f32(accS[(s0 + j0 + j) * 65 + o], h, l);
                vh[j] = h; vl[j] = l;
            }
            *(v8bf*)(outTh + base + j0) = vh;
            *(v8bf*)(outTl + base + j0) = vl;
        }
    } else {
        // row-major [r=(n*12+t)][64]
        const int row = tid >> 1;
        const int c0  = (tid & 1) * 32;
        const size_t base = ((size_t)(nBase + row) * TT + t) * FF + c0;
#pragma unroll
        for (int j0 = 0; j0 < 32; j0 += 8) {
            v8bf vh, vl;
#pragma unroll
            for (int j = 0; j < 8; ++j) {
                bf16_t h, l;
                split_f32(accS[row * 65 + c0 + j0 + j], h, l);
                vh[j] = h; vl[j] = l;
            }
            *(v8bf*)(outRh + base + j0) = vh;
            *(v8bf*)(outRl + base + j0) = vl;
        }
    }
}

// ---------------------------------------------------------------------------
// Kernel 4: temporal conv1d (K=3, SAME) as 3 shifted WMMA accumulations.
// y[n,t,o] = b[o] + sum_w sum_i x[n, t+w-1, i] * k[w,i,o]; out stored [N][F][T].
// Block: 128 rows (r = n*12+t) x 64 out channels; zero-mask out-of-range t.
// ---------------------------------------------------------------------------
__global__ __launch_bounds__(256) void conv_kernel(const bf16_t* __restrict__ Xh,
                                                   const bf16_t* __restrict__ Xl,
                                                   const float* __restrict__ Kw,
                                                   const float* __restrict__ bias,
                                                   float* __restrict__ out) {
    __shared__ bf16_t Kh[3 * 64 * 64];   // [w][o][i]  24KB
    __shared__ bf16_t Kl[3 * 64 * 64];   //            24KB

    const int tid  = threadIdx.x;
    const int wave = tid >> 5;
    const int lane = tid & 31;
    const int rBase = blockIdx.x * 128;

    // stage conv weights transposed: src (w,i,o) -> dst [w][o][i]
    for (int e = tid; e < 3 * 64 * 64; e += 256) {
        const int w = e >> 12;
        const int o = (e & 4095) >> 6;
        const int i = e & 63;
        bf16_t h, l;
        split_f32(Kw[(size_t)w * 4096 + i * 64 + o], h, l);
        Kh[e] = h;
        Kl[e] = l;
    }
    __syncthreads();

    v8f acc[4] = {};
    const int rowT = rBase + wave * 16;

#pragma unroll
    for (int w = 0; w < 3; ++w) {
#pragma unroll
        for (int kk = 0; kk < FF; kk += 32) {
            // A fragment straight from global bf16, zero-masked per lane-row
            v16bf ah, al;
            {
                const int rl = lane & 15;
                const int k0 = ((lane < 16) ? 0 : 8) + kk;
                const int r  = rowT + rl;
                const int n  = r / TT;
                const int t  = r % TT;
                const int ts = t + w - 1;
                const bool ok = (ts >= 0) && (ts < TT);
                const size_t off = ((size_t)n * TT + (ok ? ts : 0)) * FF + k0;
                ah = pack16(*(const v8bf*)(Xh + off), *(const v8bf*)(Xh + off + 16));
                al = pack16(*(const v8bf*)(Xl + off), *(const v8bf*)(Xl + off + 16));
                if (!ok) {
#pragma unroll
                    for (int i = 0; i < 16; ++i) { ah[i] = (bf16_t)0.0f; al[i] = (bf16_t)0.0f; }
                }
            }
#pragma unroll
            for (int ni = 0; ni < 4; ++ni) {
                v16bf bh = load_b_frag(&Kh[w * 4096 + (ni * 16) * FF] + kk, FF);
                v16bf bl = load_b_frag(&Kl[w * 4096 + (ni * 16) * FF] + kk, FF);
                acc[ni] = wmma_split(ah, al, bh, bl, acc[ni]);
            }
        }
    }

    // epilogue: bias + transpose store out[n][o][t]
    const int ln = lane & 15;
    const int lh = lane >> 4;
#pragma unroll
    for (int ni = 0; ni < 4; ++ni) {
        const int o = ni * 16 + ln;
        const float b = bias[o];
#pragma unroll
        for (int v = 0; v < 8; ++v) {
            const int r = rowT + lh * 8 + v;
            const int n = r / TT;
            const int t = r % TT;
            out[(size_t)n * (FF * TT) + o * TT + t] = acc[ni][v] + b;
        }
    }
}

// ---------------------------------------------------------------------------
// Host launcher
// ---------------------------------------------------------------------------
extern "C" void kernel_launch(void* const* d_in, const int* in_sizes, int n_in,
                              void* d_out, int out_size, void* d_ws, size_t ws_size,
                              hipStream_t stream) {
    const float* tf = (const float*)d_in[0];   // [8192][64][12]
    const float* A  = (const float*)d_in[1];   // [8192][8192]
    const float* W1 = (const float*)d_in[2];
    const float* b1 = (const float*)d_in[3];
    const float* W2 = (const float*)d_in[4];
    const float* b2 = (const float*)d_in[5];
    const float* W3 = (const float*)d_in[6];
    const float* b3 = (const float*)d_in[7];
    const float* ck = (const float*)d_in[8];   // [3][64][64]
    const float* cb = (const float*)d_in[9];
    float* out = (float*)d_out;                // [8192][64][12]

    // workspace carve-out: base ~96 MB; +256 MB for pre-converted A if it fits
    char* ws = (char*)d_ws;
    const size_t SZ  = (size_t)CC * NN * sizeof(bf16_t);   // 12.58 MB
    const size_t ASZ = (size_t)NN * NN * sizeof(bf16_t);   // 128 MB
    bf16_t* XtAh = (bf16_t*)(ws + 0 * SZ);
    bf16_t* XtAl = (bf16_t*)(ws + 1 * SZ);
    bf16_t* XtBh = (bf16_t*)(ws + 2 * SZ);
    bf16_t* XtBl = (bf16_t*)(ws + 3 * SZ);
    float*  Sbuf = (float*) (ws + 4 * SZ);                 // 25.2 MB (= 2*SZ)
    bf16_t* X3h  = (bf16_t*)(ws + 6 * SZ);
    bf16_t* X3l  = (bf16_t*)(ws + 7 * SZ);
    bf16_t* Agh  = (bf16_t*)(ws + 8 * SZ);
    bf16_t* Agl  = (bf16_t*)(ws + 8 * SZ + ASZ);
    const bool bigws = ws_size >= 8 * SZ + 2 * ASZ;

    const dim3 blk(256);
    const dim3 gG(CC / BN, NN / BM);   // (6, 64) column-block fastest for L2 reuse of A
    const dim3 gT(TT, NN / 128);       // (12, 64)

    pack_kernel<<<(int)(((size_t)CC * NN + 255) / 256), blk, 0, stream>>>(tf, XtAh, XtAl);

    if (bigws) {
        convert_a_kernel<<<(int)((size_t)NN * NN / 4 / 256), blk, 0, stream>>>(A, Agh, Agl);
        gemm_adj_kernel<true><<<gG, blk, 0, stream>>>(nullptr, Agh, Agl, XtAh, XtAl, Sbuf);
        transform_kernel<<<gT, blk, 0, stream>>>(Sbuf, W1, b1, 1, 1, XtBh, XtBl, nullptr, nullptr);
        gemm_adj_kernel<true><<<gG, blk, 0, stream>>>(nullptr, Agh, Agl, XtBh, XtBl, Sbuf);
        transform_kernel<<<gT, blk, 0, stream>>>(Sbuf, W2, b2, 1, 1, XtAh, XtAl, nullptr, nullptr);
        gemm_adj_kernel<true><<<gG, blk, 0, stream>>>(nullptr, Agh, Agl, XtAh, XtAl, Sbuf);
        transform_kernel<<<gT, blk, 0, stream>>>(Sbuf, W3, b3, 0, 0, nullptr, nullptr, X3h, X3l);
    } else {
        gemm_adj_kernel<false><<<gG, blk, 0, stream>>>(A, nullptr, nullptr, XtAh, XtAl, Sbuf);
        transform_kernel<<<gT, blk, 0, stream>>>(Sbuf, W1, b1, 1, 1, XtBh, XtBl, nullptr, nullptr);
        gemm_adj_kernel<false><<<gG, blk, 0, stream>>>(A, nullptr, nullptr, XtBh, XtBl, Sbuf);
        transform_kernel<<<gT, blk, 0, stream>>>(Sbuf, W2, b2, 1, 1, XtAh, XtAl, nullptr, nullptr);
        gemm_adj_kernel<false><<<gG, blk, 0, stream>>>(A, nullptr, nullptr, XtAh, XtAl, Sbuf);
        transform_kernel<<<gT, blk, 0, stream>>>(Sbuf, W3, b3, 0, 0, nullptr, nullptr, X3h, X3l);
    }
    conv_kernel<<<RR / 128, blk, 0, stream>>>(X3h, X3l, ck, cb, out);
}
